// Informer_71957882077322
// MI455X (gfx1250) — compile-verified, matrix operand order checked by reading, not run on hardware
//
#include <hip/hip_runtime.h>
#include <cmath>
#include <cstdint>
#include <cstddef>

// ---------------------------------------------------------------------------
// Types for CDNA5 WMMA (wave32, gfx1250)
// ---------------------------------------------------------------------------
typedef __bf16 bf16;
typedef __bf16 v16bf __attribute__((ext_vector_type(16)));
typedef float  v8f   __attribute__((ext_vector_type(8)));
typedef unsigned int v4u __attribute__((ext_vector_type(4)));

union Frag16 {             // 16 bf16 = 32 bytes = 8 VGPRs per lane
  v16bf v;
  v4u   u[2];
};

// ---------------------------------------------------------------------------
// Model constants
// ---------------------------------------------------------------------------
#define BN_   16
#define L0_   512
#define DM_   512
#define NH_   8
#define DH_   64
#define DFF_  2048
#define FACT_ 5
#define EPS_  1e-5f

// GEMM tile config: 128x128x32, 256 threads = 8 wave32 (2 x 4 wave grid)
#define TBM 128
#define TBN 128
#define TBK 32

// ---------------------------------------------------------------------------
// WMMA GEMM:  C[M,N] (f32) = A[M,K] (bf16) @ Bt[N,K] (bf16, pre-transposed)
// REQUIRES: M % 128 == 0, N % 128 == 0, K % 32 == 0 (true for every call here)
// A row remap supports the distill conv: arow = (m/Lout)*Lpad + m%Lout + rowOff
// flags: bit0 = relu, bit1 = accumulate into C, bit2 = add bias[n]
// Double-buffered LDS + software-pipelined global loads: 1 barrier / K-step.
// ---------------------------------------------------------------------------
__global__ __launch_bounds__(256)
void wmma_gemm_bf16(const bf16* __restrict__ A, const bf16* __restrict__ Bt,
                    const float* __restrict__ bias, float* C,
                    int M, int N, int K, int Lout, int Lpad, int rowOff,
                    int flags)
{
  __shared__ bf16 lA[2][TBM * TBK];   // lA[buf][row*32 + k]
  __shared__ bf16 lB[2][TBN * TBK];   // lB[buf][col*32 + k]

  const int tid  = threadIdx.x;
  const int bm   = blockIdx.y;
  const int bn   = blockIdx.x;
  const int lane = tid & 31;
  const int wave = tid >> 5;
  const int wm   = wave & 1;   // 2 waves along M  (64 rows each)
  const int wn   = wave >> 1;  // 4 waves along N  (32 cols each)

  v8f acc[4][2];
#pragma unroll
  for (int i = 0; i < 4; ++i)
#pragma unroll
    for (int j = 0; j < 2; ++j) acc[i][j] = {};

  const int laneM = lane & 15;
  const int kSel  = (lane < 16) ? 0 : 8;   // ISA 16-bit fragment K layout

  // Staging mapping: thread -> (row/col = tid>>1, 16 bf16 at k0=(tid&1)*16)
  const int sRow = tid >> 1;
  const int sK0  = (tid & 1) * 16;
  const int gmA  = bm * TBM + sRow;
  const size_t arowA =
      (size_t)(gmA / Lout) * (size_t)Lpad + (size_t)(gmA % Lout) + (size_t)rowOff;
  const bf16* aPtr = A  + arowA * (size_t)K + sK0;                       // + kt
  const bf16* bPtr = Bt + (size_t)(bn * TBN + sRow) * (size_t)K + sK0;   // + kt
  const int lOff = sRow * TBK + sK0;

  // ---- prologue: stage tile 0 into buffer 0 ----
  {
    v4u a0 = ((const v4u*)aPtr)[0], a1 = ((const v4u*)aPtr)[1];
    v4u b0 = ((const v4u*)bPtr)[0], b1 = ((const v4u*)bPtr)[1];
    *(v4u*)(&lA[0][lOff])     = a0;
    *(v4u*)(&lA[0][lOff + 8]) = a1;
    *(v4u*)(&lB[0][lOff])     = b0;
    *(v4u*)(&lB[0][lOff + 8]) = b1;
  }
  __syncthreads();

  int buf = 0;
  for (int kt = 0; kt < K; kt += TBK) {
    const bool hasNext = (kt + TBK) < K;
    v4u a0, a1, b0, b1;
    if (hasNext) {                    // pipelined global loads for next tile
      const bf16* an = aPtr + kt + TBK;
      const bf16* bp = bPtr + kt + TBK;
      a0 = ((const v4u*)an)[0]; a1 = ((const v4u*)an)[1];
      b0 = ((const v4u*)bp)[0]; b1 = ((const v4u*)bp)[1];
      if (kt + 2 * TBK < K) {         // CDNA5 global_prefetch_b8, one more ahead
        __builtin_prefetch((const void*)(aPtr + kt + 2 * TBK), 0, 0);
        __builtin_prefetch((const void*)(bPtr + kt + 2 * TBK), 0, 0);
      }
    }

    // ---- build WMMA fragments from current buffer (ds_load_b128 x2 each) ----
    Frag16 afr[4], bfr[2];
#pragma unroll
    for (int mi = 0; mi < 4; ++mi) {
      const bf16* p = &lA[buf][(wm * 64 + mi * 16 + laneM) * TBK + kSel];
      afr[mi].u[0] = *(const v4u*)p;
      afr[mi].u[1] = *(const v4u*)(p + 16);
    }
#pragma unroll
    for (int ni = 0; ni < 2; ++ni) {
      const bf16* p = &lB[buf][(wn * 32 + ni * 16 + laneM) * TBK + kSel];
      bfr[ni].u[0] = *(const v4u*)p;
      bfr[ni].u[1] = *(const v4u*)(p + 16);
    }

    // ---- 8 WMMA 16x16x32 ops per wave per K-step ----
#pragma unroll
    for (int mi = 0; mi < 4; ++mi)
#pragma unroll
      for (int ni = 0; ni < 2; ++ni)
        acc[mi][ni] = __builtin_amdgcn_wmma_f32_16x16x32_bf16(
            false, afr[mi].v, false, bfr[ni].v,
            (short)0, acc[mi][ni], false, false);

    if (hasNext) {                    // stage next tile into the other buffer
      *(v4u*)(&lA[buf ^ 1][lOff])     = a0;
      *(v4u*)(&lA[buf ^ 1][lOff + 8]) = a1;
      *(v4u*)(&lB[buf ^ 1][lOff])     = b0;
      *(v4u*)(&lB[buf ^ 1][lOff + 8]) = b1;
      __syncthreads();
    }
    buf ^= 1;
  }

  // ---- epilogue: ISA C/D layout (vgpr r: lanes0-15 -> M=r, lanes16-31 -> M=r+8)
  const bool relu  = flags & 1;
  const bool accum = (flags >> 1) & 1;
  const bool hasb  = (flags >> 2) & 1;
  const int rowBase = bm * TBM + wm * 64 + ((lane < 16) ? 0 : 8);
  const int colBase = bn * TBN + wn * 32 + laneM;
#pragma unroll
  for (int mi = 0; mi < 4; ++mi)
#pragma unroll
    for (int ni = 0; ni < 2; ++ni) {
      const int gn = colBase + ni * 16;
      const float bb = hasb ? bias[gn] : 0.f;
#pragma unroll
      for (int r = 0; r < 8; ++r) {
        const int gm = rowBase + mi * 16 + r;
        float v = acc[mi][ni][r] + bb;
        if (accum) v += C[(size_t)gm * N + gn];
        if (relu)  v = fmaxf(v, 0.f);
        C[(size_t)gm * N + gn] = v;
      }
    }
}

// ---------------------------------------------------------------------------
// Elementwise / helper kernels
// ---------------------------------------------------------------------------
__global__ void f2bf_kernel(const float* __restrict__ x, bf16* __restrict__ y,
                            size_t n) {
  size_t i = (size_t)blockIdx.x * 256 + threadIdx.x;
  if (i < n) y[i] = (bf16)x[i];
}

// Bt[n*K + k] = W[k*N + n]  (weight transpose + bf16 cast, once per launch)
__global__ void f2bf_tr_kernel(const float* __restrict__ w, bf16* __restrict__ bt,
                               int K, int N) {
  int idx = blockIdx.x * 256 + threadIdx.x;
  if (idx >= K * N) return;
  int k = idx % K, n = idx / K;
  bt[(size_t)n * K + k] = (bf16)w[(size_t)k * N + n];
}

__global__ void posenc_kernel(const float* __restrict__ x, float* __restrict__ o,
                              int L, int D) {
  size_t idx = (size_t)blockIdx.x * 256 + threadIdx.x;
  size_t total = (size_t)BN_ * L * D;
  if (idx >= total) return;
  int c = (int)(idx % D);
  int l = (int)((idx / D) % L);
  int i2 = (c >> 1) << 1;
  float freq = expf(-(float)i2 * 9.210340371976184f / (float)D); // ln(1e4)
  float a = (float)l * freq;
  float pe = (c & 1) ? cosf(a) : sinf(a);
  o[idx] = x[idx] + pe;
}

__global__ void add_kernel(const float* __restrict__ a, const float* __restrict__ b,
                           float* __restrict__ o, size_t n) {
  size_t i = (size_t)blockIdx.x * 256 + threadIdx.x;
  if (i < n) o[i] = a[i] + b[i];
}

// y = LayerNorm(x + r), one 512-wide row per block (256 threads x 2 elems)
__global__ __launch_bounds__(256)
void add_ln_kernel(const float* __restrict__ x, const float* __restrict__ r,
                   float* __restrict__ o, int D) {
  const int row = blockIdx.x;
  const int tid = threadIdx.x;
  const float* xr = x + (size_t)row * D;
  const float* rr = r + (size_t)row * D;
  float v0 = xr[tid] + rr[tid];
  float v1 = xr[tid + 256] + rr[tid + 256];
  __shared__ float sm[256];
  sm[tid] = v0 + v1;
  __syncthreads();
  for (int st = 128; st > 0; st >>= 1) {
    if (tid < st) sm[tid] += sm[tid + st];
    __syncthreads();
  }
  float mean = sm[0] / (float)D;
  __syncthreads();
  float d0 = v0 - mean, d1 = v1 - mean;
  sm[tid] = d0 * d0 + d1 * d1;
  __syncthreads();
  for (int st = 128; st > 0; st >>= 1) {
    if (tid < st) sm[tid] += sm[tid + st];
    __syncthreads();
  }
  float inv = rsqrtf(sm[0] / (float)D + EPS_);
  o[(size_t)row * D + tid]       = d0 * inv;
  o[(size_t)row * D + tid + 256] = d1 * inv;
}

// meanv[b, c] = mean_s V[b, s, c]
__global__ void mean_rows_kernel(const float* __restrict__ V, float* __restrict__ mv,
                                 int S, int D) {
  int idx = blockIdx.x * 256 + threadIdx.x;
  if (idx >= BN_ * D) return;
  int c = idx % D, b = idx / D;
  float s = 0.f;
  for (int r = 0; r < S; ++r) s += V[((size_t)(b * S + r)) * D + c];
  mv[idx] = s / (float)S;
}

__global__ void bcast_ctx_kernel(const float* __restrict__ mv, float* __restrict__ ctx,
                                 int L, int D) {
  size_t idx = (size_t)blockIdx.x * 256 + threadIdx.x;
  size_t total = (size_t)BN_ * L * D;
  if (idx >= total) return;
  int c = (int)(idx % D);
  int b = (int)(idx / ((size_t)L * D));
  ctx[idx] = mv[b * D + c];
}

__device__ __forceinline__ unsigned lcg_hash(int l, int j, unsigned seed) {
  unsigned h = (unsigned)l * 131071u + (unsigned)j * 2654435761u + seed * 97u + 12582917u;
  h ^= h >> 13; h *= 0x5bd1e995u; h ^= h >> 15;
  return h;
}

// ProbSparse: M-score over Upart sampled keys, iterative top-u per (b,h)
__global__ __launch_bounds__(256)
void topk_kernel(const float* __restrict__ Q, const float* __restrict__ K,
                 int* __restrict__ top, int Lq, int S, int Upart, int u,
                 unsigned seed) {
  const int bh = blockIdx.x;
  const int b = bh / NH_, h = bh % NH_;
  __shared__ float Ms[512];
  __shared__ float rv[256];
  __shared__ int   ri[256];
  for (int l = threadIdx.x; l < Lq; l += 256) {
    const float* q = Q + ((size_t)(b * Lq + l)) * DM_ + h * DH_;
    float mx = -1e30f, sm = 0.f;
    for (int j = 0; j < Upart; ++j) {
      int s = (int)(lcg_hash(l, j, seed) % (unsigned)S);
      const float* kk = K + ((size_t)(b * S + s)) * DM_ + h * DH_;
      float d = 0.f;
#pragma unroll 8
      for (int c = 0; c < DH_; ++c) d += q[c] * kk[c];
      mx = fmaxf(mx, d);
      sm += d;
    }
    Ms[l] = mx - sm / (float)S;   // reference: max - sum/L_K
  }
  __syncthreads();
  for (int it = 0; it < u; ++it) {
    float bv = -1e30f; int bi = 0;
    for (int l = threadIdx.x; l < Lq; l += 256)
      if (Ms[l] > bv) { bv = Ms[l]; bi = l; }
    rv[threadIdx.x] = bv; ri[threadIdx.x] = bi;
    __syncthreads();
    for (int st = 128; st > 0; st >>= 1) {
      if (threadIdx.x < st && rv[threadIdx.x + st] > rv[threadIdx.x]) {
        rv[threadIdx.x] = rv[threadIdx.x + st];
        ri[threadIdx.x] = ri[threadIdx.x + st];
      }
      __syncthreads();
    }
    if (threadIdx.x == 0) { top[bh * 64 + it] = ri[0]; Ms[ri[0]] = -1e30f; }
    __syncthreads();
  }
}

// Softmax(QK^T/sqrt(dh)) @ V for the selected queries; one block per (b,h,u)
__global__ __launch_bounds__(64)
void attn_update_kernel(const float* __restrict__ Q, const float* __restrict__ K,
                        const float* __restrict__ V, const int* __restrict__ top,
                        float* __restrict__ ctx, int Lq, int S, int u) {
  const int gid = blockIdx.x;
  const int iu = gid % u;
  const int bh = gid / u;
  const int b = bh / NH_, h = bh % NH_;
  const int l = top[bh * 64 + iu];
  const int t = threadIdx.x;  // 0..63 == head dim
  __shared__ float sc[512];
  __shared__ float red[64];
  const float* q = Q + ((size_t)(b * Lq + l)) * DM_ + h * DH_;
  const float scale = 0.125f;  // 1/sqrt(64)
  for (int s = t; s < S; s += 64) {
    const float* kk = K + ((size_t)(b * S + s)) * DM_ + h * DH_;
    float d = 0.f;
#pragma unroll 8
    for (int c = 0; c < DH_; ++c) d += q[c] * kk[c];
    sc[s] = d * scale;
  }
  __syncthreads();
  float mx = -1e30f;
  for (int s = t; s < S; s += 64) mx = fmaxf(mx, sc[s]);
  red[t] = mx; __syncthreads();
  for (int st = 32; st > 0; st >>= 1) {
    if (t < st) red[t] = fmaxf(red[t], red[t + st]);
    __syncthreads();
  }
  mx = red[0];
  __syncthreads();
  float sum = 0.f;
  for (int s = t; s < S; s += 64) { float e = expf(sc[s] - mx); sc[s] = e; sum += e; }
  red[t] = sum; __syncthreads();
  for (int st = 32; st > 0; st >>= 1) {
    if (t < st) red[t] += red[t + st];
    __syncthreads();
  }
  float inv = 1.f / red[0];
  __syncthreads();
  float acc = 0.f;
  for (int s = 0; s < S; ++s) acc += sc[s] * V[((size_t)(b * S + s)) * DM_ + h * DH_ + t];
  ctx[((size_t)(b * Lq + l)) * DM_ + h * DH_ + t] = acc * inv;
}

// Circular pad (B,L,D)->(B,L+2,D) and cast to bf16
__global__ void pad_bf16_kernel(const float* __restrict__ x, bf16* __restrict__ p,
                                int L, int D) {
  size_t idx = (size_t)blockIdx.x * 256 + threadIdx.x;
  size_t total = (size_t)BN_ * (L + 2) * D;
  if (idx >= total) return;
  int c = (int)(idx % D);
  int l = (int)((idx / D) % (L + 2));
  int b = (int)(idx / ((size_t)(L + 2) * D));
  int ls = (l == 0) ? (L - 1) : ((l == L + 1) ? 0 : l - 1);
  p[idx] = (bf16)x[((size_t)(b * L + ls)) * D + c];
}

// Per-channel mean/var over all (b,l) rows  (distill batch-norm stats)
__global__ __launch_bounds__(256)
void bn_stats_kernel(const float* __restrict__ y, float* __restrict__ mu,
                     float* __restrict__ var, int rows, int D) {
  const int c = blockIdx.x;
  float s = 0.f, s2 = 0.f;
  for (int r = threadIdx.x; r < rows; r += 256) {
    float v = y[(size_t)r * D + c];
    s += v; s2 += v * v;
  }
  __shared__ float a[256], b2[256];
  a[threadIdx.x] = s; b2[threadIdx.x] = s2;
  __syncthreads();
  for (int st = 128; st > 0; st >>= 1) {
    if (threadIdx.x < st) { a[threadIdx.x] += a[threadIdx.x + st]; b2[threadIdx.x] += b2[threadIdx.x + st]; }
    __syncthreads();
  }
  if (threadIdx.x == 0) {
    float m = a[0] / (float)rows;
    mu[c] = m;
    var[c] = b2[0] / (float)rows - m * m;
  }
}

// out[b,lo,c] = max over window of elu(norm(y[b,l,c])) ; window 3, stride 2, pad 1
__global__ void elu_maxpool_kernel(const float* __restrict__ y,
                                   const float* __restrict__ mu,
                                   const float* __restrict__ var,
                                   float* __restrict__ out,
                                   int L, int Lout, int D) {
  size_t idx = (size_t)blockIdx.x * 256 + threadIdx.x;
  size_t total = (size_t)BN_ * Lout * D;
  if (idx >= total) return;
  int c = (int)(idx % D);
  int lo = (int)((idx / D) % Lout);
  int b = (int)(idx / ((size_t)Lout * D));
  float m = mu[c];
  float iv = rsqrtf(var[c] + EPS_);
  float best = -1e30f;
  for (int w = 0; w < 3; ++w) {
    int l = 2 * lo - 1 + w;
    if (l < 0 || l >= L) continue;
    float v = (y[((size_t)(b * L + l)) * D + c] - m) * iv;
    v = (v > 0.f) ? v : (expf(v) - 1.f);
    best = fmaxf(best, v);
  }
  out[idx] = best;
}

// Pack dist_w (co,ci,3) -> 3 bf16 transposed matrices dwt[t][co*D + ci] (Bt: N=co rows, K=ci)
__global__ void pack_dw_kernel(const float* __restrict__ w, bf16* __restrict__ dwt,
                               int D) {
  int idx = blockIdx.x * 256 + threadIdx.x;
  if (idx >= D * D * 3) return;
  int t = idx % 3;
  int ci = (idx / 3) % D;
  int co = idx / (3 * D);
  dwt[(size_t)t * D * D + (size_t)co * D + ci] = (bf16)w[idx];
}

// Final head: out[m] = sum_c X[m,c] * w[c] + b0 ; one wave32 per row
__global__ __launch_bounds__(256)
void gemv_out_kernel(const float* __restrict__ X, const float* __restrict__ w,
                     const float* __restrict__ b0, float* __restrict__ o,
                     int M, int D) {
  int m = blockIdx.x * 8 + (threadIdx.x >> 5);
  int lane = threadIdx.x & 31;
  if (m >= M) return;
  float s = 0.f;
  for (int c = lane; c < D; c += 32) s += X[(size_t)m * D + c] * w[c];
  for (int off = 16; off > 0; off >>= 1) s += __shfl_down(s, off, 32);
  if (lane == 0) o[m] = s + b0[0];
}

// ---------------------------------------------------------------------------
// Host orchestration
// ---------------------------------------------------------------------------
extern "C" void kernel_launch(void* const* d_in, const int* in_sizes, int n_in,
                              void* d_out, int out_size, void* d_ws, size_t ws_size,
                              hipStream_t stream) {
  (void)in_sizes; (void)n_in; (void)out_size; (void)ws_size;

  const float* in_x   = (const float*)d_in[0];
  const float* Wq     = (const float*)d_in[1];
  const float* bq     = (const float*)d_in[2];
  const float* Wk     = (const float*)d_in[3];
  const float* bk     = (const float*)d_in[4];
  const float* Wv     = (const float*)d_in[5];
  const float* bv     = (const float*)d_in[6];
  const float* Wo     = (const float*)d_in[7];
  const float* bo     = (const float*)d_in[8];
  const float* enc_w1 = (const float*)d_in[9];
  const float* enc_b1 = (const float*)d_in[10];
  const float* enc_w2 = (const float*)d_in[11];
  const float* enc_b2 = (const float*)d_in[12];
  const float* dec_w1 = (const float*)d_in[13];
  const float* dec_b1 = (const float*)d_in[14];
  const float* dec_w2 = (const float*)d_in[15];
  const float* dec_b2 = (const float*)d_in[16];
  const float* dist_w = (const float*)d_in[17];
  const float* dist_b = (const float*)d_in[18];
  const float* out_w  = (const float*)d_in[19];
  const float* out_b  = (const float*)d_in[20];
  float* out = (float*)d_out;

  const int D = DM_, DFF = DFF_;
  const size_t M0 = (size_t)BN_ * L0_;

  // ---- workspace carve (bump allocator, 256B aligned) ----
  char* wp = (char*)d_ws;
  auto alloc = [&](size_t bytes) -> char* {
    char* p = wp;
    wp += (bytes + 255) & ~(size_t)255;
    return p;
  };
  float* xe   = (float*)alloc(M0 * D * 4);        // encoder stream (len shrinks)
  float* xd   = (float*)alloc(M0 * D * 4);        // decoder input
  float* qb   = (float*)alloc(M0 * D * 4);
  float* kb   = (float*)alloc(M0 * D * 4);
  float* vb   = (float*)alloc(M0 * D * 4);
  float* ctx  = (float*)alloc(M0 * D * 4);
  float* t1   = (float*)alloc(M0 * D * 4);
  float* t2   = (float*)alloc(M0 * D * 4);
  float* t3   = (float*)alloc(M0 * D * 4);
  float* hbuf = (float*)alloc(M0 * DFF * 4);
  float* mv   = (float*)alloc((size_t)BN_ * D * 4);
  float* muB  = (float*)alloc(D * 4);
  float* varB = (float*)alloc(D * 4);
  int*   top  = (int*)alloc((size_t)BN_ * NH_ * 64 * 4);
  bf16*  abf  = (bf16*)alloc(M0 * DFF * 2);       // activation bf16 staging
  bf16*  pbf  = (bf16*)alloc((size_t)BN_ * (L0_ + 2) * D * 2);
  bf16*  wqb  = (bf16*)alloc((size_t)D * D * 2);  // all weights pre-transposed NxK
  bf16*  wkb  = (bf16*)alloc((size_t)D * D * 2);
  bf16*  wvb  = (bf16*)alloc((size_t)D * D * 2);
  bf16*  wob  = (bf16*)alloc((size_t)D * D * 2);
  bf16*  w1e  = (bf16*)alloc((size_t)D * DFF * 2);
  bf16*  w2e  = (bf16*)alloc((size_t)DFF * D * 2);
  bf16*  w1d  = (bf16*)alloc((size_t)D * DFF * 2);
  bf16*  w2d  = (bf16*)alloc((size_t)DFF * D * 2);
  bf16*  dwt  = (bf16*)alloc((size_t)3 * D * D * 2);

  auto cvt = [&](const float* s, bf16* d, size_t n) {
    f2bf_kernel<<<(unsigned)((n + 255) / 256), 256, 0, stream>>>(s, d, n);
  };
  auto trb = [&](const float* w, bf16* bt, int K, int N) {
    f2bf_tr_kernel<<<(K * N + 255) / 256, 256, 0, stream>>>(w, bt, K, N);
  };
  auto gemm = [&](const bf16* A, const bf16* Bm, const float* bias, float* C,
                  int M, int N, int K, int Lout, int Lpad, int roff, int flags) {
    dim3 g((N + TBN - 1) / TBN, (M + TBM - 1) / TBM);
    wmma_gemm_bf16<<<g, 256, 0, stream>>>(A, Bm, bias, C, M, N, K, Lout, Lpad, roff, flags);
  };
  auto ceil_log = [](int L) { return (int)ceil(log((double)L)); };
  auto ufor = [&](int L) { int v = FACT_ * ceil_log(L); return v < L ? v : L; };

  // ---- weight conversion + transpose (every call; deterministic, cheap) ----
  trb(Wq, wqb, D, D);  trb(Wk, wkb, D, D);
  trb(Wv, wvb, D, D);  trb(Wo, wob, D, D);
  trb(enc_w1, w1e, D, DFF);  trb(enc_w2, w2e, DFF, D);
  trb(dec_w1, w1d, D, DFF);  trb(dec_w2, w2d, DFF, D);
  pack_dw_kernel<<<(D * D * 3 + 255) / 256, 256, 0, stream>>>(dist_w, dwt, D);

  // ---- ProbSparse attention: out = proj(ctx) ----
  auto attn = [&](const float* qin, const float* kvin, int Lq, int S,
                  unsigned seed, float* outp) {
    const int Mq = BN_ * Lq, Msz = BN_ * S;
    cvt(qin, abf, (size_t)Mq * D);
    gemm(abf, wqb, bq, qb, Mq, D, D, Lq, Lq, 0, 4);
    if (kvin != qin) cvt(kvin, abf, (size_t)Msz * D);
    gemm(abf, wkb, bk, kb, Msz, D, D, S, S, 0, 4);
    gemm(abf, wvb, bv, vb, Msz, D, D, S, S, 0, 4);
    mean_rows_kernel<<<(BN_ * D + 255) / 256, 256, 0, stream>>>(vb, mv, S, D);
    bcast_ctx_kernel<<<(unsigned)(((size_t)Mq * D + 255) / 256), 256, 0, stream>>>(mv, ctx, Lq, D);
    int Up = ufor(S), u = ufor(Lq);
    topk_kernel<<<BN_ * NH_, 256, 0, stream>>>(qb, kb, top, Lq, S, Up, u, seed);
    attn_update_kernel<<<BN_ * NH_ * u, 64, 0, stream>>>(qb, kb, vb, top, ctx, Lq, S, u);
    cvt(ctx, abf, (size_t)Mq * D);
    gemm(abf, wob, bo, outp, Mq, D, D, Lq, Lq, 0, 4);
  };

  auto encoder = [&](float* x, int L, unsigned seed) {
    const int M = BN_ * L;
    attn(x, x, L, L, seed, t1);
    add_ln_kernel<<<M, 256, 0, stream>>>(x, t1, t2, D);
    cvt(t2, abf, (size_t)M * D);
    gemm(abf, w1e, enc_b1, hbuf, M, DFF, D, L, L, 0, 4 | 1);
    cvt(hbuf, abf, (size_t)M * DFF);
    gemm(abf, w2e, enc_b2, t1, M, D, DFF, L, L, 0, 4);
    add_ln_kernel<<<M, 256, 0, stream>>>(t2, t1, x, D);
  };

  auto distill = [&](float* x, int L) {  // in-place; new length L/2
    const int M = BN_ * L;
    pad_bf16_kernel<<<(unsigned)(((size_t)BN_ * (L + 2) * D + 255) / 256), 256, 0, stream>>>(x, pbf, L, D);
    gemm(pbf, dwt,                     dist_b,  t1, M, D, D, L, L + 2, 0, 4);
    gemm(pbf, dwt + (size_t)D * D,     nullptr, t1, M, D, D, L, L + 2, 1, 2);
    gemm(pbf, dwt + 2 * (size_t)D * D, nullptr, t1, M, D, D, L, L + 2, 2, 2);
    bn_stats_kernel<<<D, 256, 0, stream>>>(t1, muB, varB, M, D);
    const int Lout = L / 2;
    elu_maxpool_kernel<<<(unsigned)(((size_t)BN_ * Lout * D + 255) / 256), 256, 0, stream>>>(
        t1, muB, varB, x, L, Lout, D);
  };

  // ---- forward pass ----
  posenc_kernel<<<(unsigned)((M0 * D + 255) / 256), 256, 0, stream>>>(in_x, xe, L0_, D);
  posenc_kernel<<<(unsigned)((M0 * D + 255) / 256), 256, 0, stream>>>(in_x, xd, L0_, D);

  encoder(xe, 512, 0); distill(xe, 512);   // -> L=256
  encoder(xe, 256, 1); distill(xe, 256);   // -> L=128
  encoder(xe, 128, 2); distill(xe, 128);   // -> L=64

  // decoder
  attn(xd, xd, 512, 512, 3, t1);
  add_kernel<<<(unsigned)((M0 * D + 255) / 256), 256, 0, stream>>>(xd, t1, t2, M0 * D);  // out1
  attn(t2, xe, 512, 64, 4, t1);
  add_kernel<<<(unsigned)((M0 * D + 255) / 256), 256, 0, stream>>>(t2, t1, t3, M0 * D);  // out2
  cvt(t3, abf, M0 * D);
  gemm(abf, w1d, dec_b1, hbuf, (int)M0, DFF, D, 512, 512, 0, 4 | 1);
  cvt(hbuf, abf, M0 * DFF);
  gemm(abf, w2d, dec_b2, t1, (int)M0, D, DFF, 512, 512, 0, 4);
  add_kernel<<<(unsigned)((M0 * D + 255) / 256), 256, 0, stream>>>(t1, t3, t2, M0 * D);  // out3

  gemv_out_kernel<<<(unsigned)((M0 + 7) / 8), 256, 0, stream>>>(t2, out_w, out_b, out, (int)M0, D);
}
